// Head_60086592471343
// MI455X (gfx1250) — matrix operand order repaired and checked by reading, not compiled, not run on hardware
//
#include <hip/hip_runtime.h>

// Problem constants (match reference)
#define Bsz 4
#define Tsz 4096
#define Csz 1024
#define HSz 64
#define NQT (Tsz / 16)    // 256 query tiles per batch
#define KCH 512           // keys per split-K chunk
#define NCH 8             // max chunks per q-tile (T / KCH)

typedef __attribute__((ext_vector_type(16))) __bf16 v16bf;
typedef __attribute__((ext_vector_type(8)))  float  v8f;

// ---------------------------------------------------------------------------
// Kernel 0: weight prep.  W[C,HS] f32 -> Wt[3][HS][C] bf16 (transposed) so
// every projection B-fragment is one contiguous 32-byte load.
// ---------------------------------------------------------------------------
__global__ __launch_bounds__(256) void w_prep_kernel(
    const float* __restrict__ Wk, const float* __restrict__ Wq,
    const float* __restrict__ Wv, __bf16* __restrict__ Wt)
{
    const int idx = blockIdx.x * 256 + threadIdx.x;   // 3*HS*C elements
    const int c   = idx % Csz;
    const int n   = (idx / Csz) % HSz;
    const int mat = idx / (Csz * HSz);
    const float* W = (mat == 0) ? Wk : (mat == 1) ? Wq : Wv;
    Wt[idx] = (__bf16)W[(size_t)c * HSz + n];
}

// ---------------------------------------------------------------------------
// Kernel 1: QKV projection.  x[B,T,C] @ W[C,HS] + b -> bf16 buffers.
//   Kb, Qb row-major [B,T,HS]; V transposed Vt[B,HS,T]; Q pre-scaled 0.125.
// One wave = one 16-row M-tile of one matrix, all 4 N-tiles.
// ---------------------------------------------------------------------------
__global__ __launch_bounds__(256) void qkv_proj_kernel(
    const float* __restrict__ x,
    const __bf16* __restrict__ Wt,
    const float* __restrict__ bk, const float* __restrict__ bq,
    const float* __restrict__ bv,
    __bf16* __restrict__ Kb, __bf16* __restrict__ Qb, __bf16* __restrict__ Vt)
{
    const int wid  = blockIdx.x * 8 + (threadIdx.x >> 5);
    const int lane = threadIdx.x & 31;
    const int mat  = wid % 3;                // 0=K, 1=Q, 2=V
    const int tile = wid / 3;
    const int b    = tile / NQT;
    const int t0   = (tile % NQT) * 16;

    const float* bias = (mat == 0) ? bk : (mat == 1) ? bq : bv;

    const int half = lane >> 4;
    const int l16  = lane & 15;

    v8f acc[4];
    v8f zero = {};
    #pragma unroll
    for (int nt = 0; nt < 4; ++nt) acc[nt] = zero;

    const float*  xrow  = x + ((size_t)b * Tsz + t0 + l16) * Csz;
    const __bf16* wbase = Wt + ((size_t)mat * HSz + l16) * Csz + 16 * half;

    for (int kk = 0; kk < Csz; kk += 32) {
        union { v16bf v; } a;
        const float* s1 = xrow + kk + 16 * half;
        const float* s2 = xrow + kk + 16 * half + 8;
        #pragma unroll
        for (int i = 0; i < 8; ++i) {
            a.v[i]     = (__bf16)s1[i];
            a.v[8 + i] = (__bf16)s2[i];
        }
        #pragma unroll
        for (int nt = 0; nt < 4; ++nt) {
            v16bf bb = *(const v16bf*)(wbase + (size_t)(nt * 16) * Csz + kk);
            acc[nt] = __builtin_amdgcn_wmma_f32_16x16x32_bf16(
                false, a.v, false, bb, (short)0, acc[nt], false, false);
        }
    }

    #pragma unroll
    for (int nt = 0; nt < 4; ++nt) {
        const int   col = nt * 16 + l16;
        const float bv_ = bias[col];
        #pragma unroll
        for (int r = 0; r < 8; ++r) {
            const int m = r + 8 * half;
            float val = acc[nt][r] + bv_;
            if (mat == 1) val *= 0.125f;
            if (mat == 2) {
                Vt[((size_t)b * HSz + col) * Tsz + t0 + m] = (__bf16)val;
            } else if (mat == 1) {
                Qb[((size_t)b * Tsz + t0 + m) * HSz + col] = (__bf16)val;
            } else {
                Kb[((size_t)b * Tsz + t0 + m) * HSz + col] = (__bf16)val;
            }
        }
    }
}

// ---------------------------------------------------------------------------
// Kernel 2: split-K causal flash attention (partials).
// Grid: (b, qt, chunk) rectangular; one wave = one 512-key chunk of one
// 16-query tile.  Slots past the diagonal exit immediately (wave-uniform).
// Inner loop identical WMMA pipeline; epilogue stores unnormalized O and
// per-row (m, l) to workspace for the combine kernel.
// ---------------------------------------------------------------------------
__global__ __launch_bounds__(256) void attn_partial_kernel(
    const __bf16* __restrict__ Kb, const __bf16* __restrict__ Qb,
    const __bf16* __restrict__ Vt,
    float* __restrict__ Opart, float* __restrict__ Mpart,
    float* __restrict__ Lpart)
{
    __shared__ __align__(16) __bf16 lds[8][16 * 64];   // per-wave 16x64 P tile

    const int wv   = threadIdx.x >> 5;
    const int lane = threadIdx.x & 31;
    const int wid  = blockIdx.x * 8 + wv;      // 0 .. 8191
    const int c    = wid & (NCH - 1);
    const int qt   = (wid >> 3) & (NQT - 1);
    const int b    = wid >> 11;
    const int t0   = qt * 16;
    const int sbeg = c * KCH;
    const int send0 = t0 + 16;
    const int send  = (sbeg + KCH < send0) ? sbeg + KCH : send0;
    if (sbeg >= send0) return;                 // wave-uniform: idle slot

    const int half = lane >> 4;
    const int l16  = lane & 15;
    v8f zero = {};

    // --- Q A-fragments (HS=64 -> two 16x32 fragments) ---
    v16bf qa[2];
    {
        const __bf16* qrow = Qb + ((size_t)b * Tsz + t0 + l16) * HSz;
        #pragma unroll
        for (int kb = 0; kb < 2; ++kb) {
            union { uint4 q[2]; v16bf v; } u;
            u.q[0] = *(const uint4*)(qrow + 32 * kb + 16 * half);
            u.q[1] = *(const uint4*)(qrow + 32 * kb + 16 * half + 8);
            qa[kb] = u.v;
        }
    }

    v8f o[4];
    #pragma unroll
    for (int nt = 0; nt < 4; ++nt) o[nt] = zero;
    float mrow[8], lrow[8];
    #pragma unroll
    for (int r = 0; r < 8; ++r) { mrow[r] = -__builtin_inff(); lrow[r] = 0.0f; }

    for (int s0 = sbeg; s0 < send; s0 += 64) {
        const __bf16* kblk = Kb + ((size_t)b * Tsz + s0) * HSz;
        __builtin_prefetch(kblk + (size_t)64 * HSz + (size_t)l16 * HSz, 0, 1);

        // ---- S = Q K^T : four 16x16 key sub-tiles ----
        v8f sa[4];
        #pragma unroll
        for (int j = 0; j < 4; ++j) sa[j] = zero;
        #pragma unroll
        for (int kb = 0; kb < 2; ++kb) {
            const __bf16* kbase = kblk + 32 * kb + 16 * half;
            #pragma unroll
            for (int j = 0; j < 4; ++j) {
                v16bf kf = *(const v16bf*)(kbase + (size_t)(16 * j + l16) * HSz);
                sa[j] = __builtin_amdgcn_wmma_f32_16x16x32_bf16(
                    false, qa[kb], false, kf, (short)0, sa[j], false, false);
            }
        }

        // ---- causal mask (per-lane selects; EXEC stays full) ----
        if (s0 + 63 > t0) {
            #pragma unroll
            for (int j = 0; j < 4; ++j)
                #pragma unroll
                for (int r = 0; r < 8; ++r) {
                    const int qg = t0 + r + 8 * half;
                    sa[j][r] = (s0 + 16 * j + l16 > qg) ? -__builtin_inff()
                                                        : sa[j][r];
                }
        }

        // ---- online softmax; row r lives across a 16-lane group ----
        float alpha[8];
        #pragma unroll
        for (int r = 0; r < 8; ++r) {
            float mx = fmaxf(fmaxf(sa[0][r], sa[1][r]),
                             fmaxf(sa[2][r], sa[3][r]));
            mx = fmaxf(mx, __shfl_xor(mx, 8, 32));
            mx = fmaxf(mx, __shfl_xor(mx, 4, 32));
            mx = fmaxf(mx, __shfl_xor(mx, 2, 32));
            mx = fmaxf(mx, __shfl_xor(mx, 1, 32));
            const float mnew = fmaxf(mrow[r], mx);
            const float al   = __expf(mrow[r] - mnew);
            float rs = 0.0f;
            #pragma unroll
            for (int j = 0; j < 4; ++j) {
                const float p = __expf(sa[j][r] - mnew);
                sa[j][r] = p;
                rs += p;
            }
            rs += __shfl_xor(rs, 8, 32);
            rs += __shfl_xor(rs, 4, 32);
            rs += __shfl_xor(rs, 2, 32);
            rs += __shfl_xor(rs, 1, 32);
            lrow[r]  = lrow[r] * al + rs;
            mrow[r]  = mnew;
            alpha[r] = al;
        }

        #pragma unroll
        for (int nt = 0; nt < 4; ++nt)
            #pragma unroll
            for (int r = 0; r < 8; ++r) o[nt][r] *= alpha[r];

        // ---- P relayout C->A via per-wave LDS (row-major 16x64 bf16) ----
        __bf16* pl = lds[wv];
        #pragma unroll
        for (int j = 0; j < 4; ++j)
            #pragma unroll
            for (int r = 0; r < 8; ++r)
                pl[(r + 8 * half) * 64 + 16 * j + l16] = (__bf16)sa[j][r];
        asm volatile("s_wait_dscnt 0" ::: "memory");

        v16bf pa[2];
        {
            const unsigned int* p32 = (const unsigned int*)(pl + (size_t)l16 * 64);
            #pragma unroll
            for (int kb = 0; kb < 2; ++kb) {
                union { unsigned int w[8]; v16bf v; } u;
                #pragma unroll
                for (int j = 0; j < 4; ++j) {
                    u.w[j]     = p32[16 * kb + 8 * half + j];
                    u.w[4 + j] = p32[16 * kb + 8 * half + 4 + j];
                }
                pa[kb] = u.v;
            }
        }

        // ---- O += P V : contiguous V^T B-fragments ----
        #pragma unroll
        for (int kb = 0; kb < 2; ++kb) {
            #pragma unroll
            for (int nt = 0; nt < 4; ++nt) {
                const __bf16* vb = Vt + ((size_t)b * HSz + nt * 16 + l16) * Tsz
                                   + s0 + 32 * kb + 16 * half;
                v16bf vf = *(const v16bf*)vb;
                o[nt] = __builtin_amdgcn_wmma_f32_16x16x32_bf16(
                    false, pa[kb], false, vf, (short)0, o[nt], false, false);
            }
        }
    }

    // ---- store partials: unnormalized O, row max m, row sum l ----
    const size_t pbase = (((size_t)(b * NQT + qt)) * NCH + c) * 16;
    #pragma unroll
    for (int nt = 0; nt < 4; ++nt)
        #pragma unroll
        for (int r = 0; r < 8; ++r)
            Opart[(pbase + r + 8 * half) * HSz + nt * 16 + l16] = o[nt][r];
    if (l16 == 0) {
        #pragma unroll
        for (int r = 0; r < 8; ++r) {
            Mpart[pbase + r + 8 * half] = mrow[r];
            Lpart[pbase + r + 8 * half] = lrow[r];
        }
    }
}

// ---------------------------------------------------------------------------
// Kernel 3: combine split-K partials.  One wave per (b, qt): log-sum-exp
// merge of up to 8 chunk partials, write fp32 output [B,T,HS].
// Lane handles columns (lane, lane+32).
// ---------------------------------------------------------------------------
__global__ __launch_bounds__(256) void attn_combine_kernel(
    const float* __restrict__ Opart, const float* __restrict__ Mpart,
    const float* __restrict__ Lpart, float* __restrict__ out)
{
    const int wv   = threadIdx.x >> 5;
    const int lane = threadIdx.x & 31;
    const int wid  = blockIdx.x * 8 + wv;      // 0 .. 1023
    const int b    = wid / NQT;
    const int qt   = wid % NQT;
    const int t0   = qt * 16;
    const int nch  = (t0 + 16 + KCH - 1) / KCH;

    const size_t baseML = ((size_t)(b * NQT + qt)) * NCH * 16;
    const size_t baseO  = baseML * HSz;

    for (int m = 0; m < 16; ++m) {
        float mstar = -__builtin_inff();
        for (int c = 0; c < nch; ++c)
            mstar = fmaxf(mstar, Mpart[baseML + c * 16 + m]);
        float denom = 0.0f, a0 = 0.0f, a1 = 0.0f;
        for (int c = 0; c < nch; ++c) {
            const float w = __expf(Mpart[baseML + c * 16 + m] - mstar);
            denom += w * Lpart[baseML + c * 16 + m];
            const float* op = Opart + baseO + (size_t)(c * 16 + m) * HSz;
            a0 += w * op[lane];
            a1 += w * op[lane + 32];
        }
        const float inv = 1.0f / denom;
        float* orow = out + ((size_t)b * Tsz + t0 + m) * HSz;
        orow[lane]      = a0 * inv;
        orow[lane + 32] = a1 * inv;
    }
}

// ---------------------------------------------------------------------------
extern "C" void kernel_launch(void* const* d_in, const int* in_sizes, int n_in,
                              void* d_out, int out_size, void* d_ws, size_t ws_size,
                              hipStream_t stream) {
    const float* x  = (const float*)d_in[0];
    const float* Wk = (const float*)d_in[1];
    const float* bk = (const float*)d_in[2];
    const float* Wq = (const float*)d_in[3];
    const float* bq = (const float*)d_in[4];
    const float* Wv = (const float*)d_in[5];
    const float* bv = (const float*)d_in[6];
    float* out = (float*)d_out;

    char* ws = (char*)d_ws;
    const size_t qkv_bytes = (size_t)Bsz * Tsz * HSz * sizeof(__bf16);  // 2 MB each
    const size_t wt_bytes  = (size_t)3 * HSz * Csz * sizeof(__bf16);    // 384 KB
    const size_t nslots    = (size_t)Bsz * NQT * NCH * 16;              // partial rows

    __bf16* Kb = (__bf16*)(ws);
    __bf16* Qb = (__bf16*)(ws + qkv_bytes);
    __bf16* Vt = (__bf16*)(ws + 2 * qkv_bytes);
    __bf16* Wt = (__bf16*)(ws + 3 * qkv_bytes);
    float*  Opart = (float*)(ws + 3 * qkv_bytes + wt_bytes);            // 32 MB
    float*  Mpart = Opart + nslots * HSz;
    float*  Lpart = Mpart + nslots;

    // kernel 0: weight transpose+convert: 3*HS*C / 256 = 768 blocks
    w_prep_kernel<<<768, 256, 0, stream>>>(Wk, Wq, Wv, Wt);
    // kernel 1: 3 * B * NQT = 3072 waves -> 384 blocks of 8 waves
    qkv_proj_kernel<<<384, 256, 0, stream>>>(x, Wt, bk, bq, bv, Kb, Qb, Vt);
    // kernel 2: B * NQT * NCH = 8192 wave slots -> 1024 blocks of 8 waves
    attn_partial_kernel<<<1024, 256, 0, stream>>>(Kb, Qb, Vt, Opart, Mpart, Lpart);
    // kernel 3: B * NQT = 1024 waves -> 128 blocks of 8 waves
    attn_combine_kernel<<<128, 256, 0, stream>>>(Opart, Mpart, Lpart, out);
}
